// FairCDSR_21998822490706
// MI455X (gfx1250) — compile-verified, matrix operand order checked by reading, not run on hardware
//
#include <hip/hip_runtime.h>
#include <hip/hip_bf16.h>

// ---------------------------------------------------------------------------
// FairCDSR triple-encoder forward for MI455X (gfx1250, wave32, WMMA).
//
// Roofline: ~184 GFLOP vs ~450MB activation traffic -> HBM-bound at 23.3TB/s
// (~20us floor). Dense projections AND attention (QK^T, PV) run through
// v_wmma_f32_16x16x32_f16 with f32 accumulation. Weights are converted to
// f16 ONCE per launch (L2-resident thereafter); Q/K/V tensors are produced
// directly in f16 by the projection GEMM epilogue, halving their HBM traffic
// and making attention staging a pure b128 copy.
// ---------------------------------------------------------------------------

typedef __attribute__((ext_vector_type(16))) _Float16 v16h;
typedef __attribute__((ext_vector_type(8)))  float    v8f;

#define Bdim   512
#define Ldim   200
#define Ddim   128
#define Hdim   2
#define DHdim  64
#define NTOK   (Bdim * Ldim)            // 102400 tokens
#define NELT   ((size_t)NTOK * Ddim)    // elements per activation tensor
#define MASKID 50000                    // ITEMNUM - 1
#define EMBSCALE 11.313708498984760f    // sqrt(128)
#define NQT    13                       // ceil(L/16) query tiles
#define WMAT   (128 * 128)              // weight matrix elements

__device__ __forceinline__ v8f wmma_f16(v16h a, v16h b, v8f c) {
  return __builtin_amdgcn_wmma_f32_16x16x32_f16(false, a, false, b,
                                                (short)0, c, false, false);
}

// ---------------------------------------------------------------------------
// One-time weight conversion f32 -> f16 (vectorized).
// ---------------------------------------------------------------------------
__global__ void wcvt_kernel(const float* __restrict__ in,
                            _Float16* __restrict__ out) {
  int i = (blockIdx.x * 256 + threadIdx.x) * 4;   // 4 elems/thread
  float4 f = *reinterpret_cast<const float4*>(in + i);
  union { _Float16 h[4]; uint2 u; } cv;
  cv.h[0] = (_Float16)f.x; cv.h[1] = (_Float16)f.y;
  cv.h[2] = (_Float16)f.z; cv.h[3] = (_Float16)f.w;
  *reinterpret_cast<uint2*>(out + i) = cv.u;
}

// ---------------------------------------------------------------------------
// Embedding gather + positional add + timeline mask.
// ---------------------------------------------------------------------------
__global__ void embed_kernel(const int* __restrict__ ids,
                             const int* __restrict__ pos,
                             const float* __restrict__ emb,
                             const float* __restrict__ pemb,
                             float* __restrict__ x,
                             float* __restrict__ tmaskf) {
  int t = blockIdx.x;          // token
  int d = threadIdx.x;         // 0..127
  int id = ids[t];
  int pp = pos[t];
  float m = (id != MASKID) ? 1.0f : 0.0f;
  float v = (emb[(size_t)id * Ddim + d] * EMBSCALE +
             pemb[(size_t)pp * Ddim + d]) * m;
  x[(size_t)t * Ddim + d] = v;
  if (d == 0) tmaskf[t] = m;
}

// ---------------------------------------------------------------------------
// LayerNorm over D=128, one wave32 per token (4 elems/lane).
// ---------------------------------------------------------------------------
__global__ void ln_kernel(const float* __restrict__ in,
                          const float* __restrict__ w,
                          const float* __restrict__ b,
                          float* __restrict__ out, int ntok) {
  int t = blockIdx.x * blockDim.y + threadIdx.y;
  if (t >= ntok) return;
  int lane = threadIdx.x;
  const float* p = in + (size_t)t * Ddim;
  float v[4];
  float s = 0.0f;
#pragma unroll
  for (int i = 0; i < 4; ++i) { v[i] = p[lane + 32 * i]; s += v[i]; }
#pragma unroll
  for (int o = 16; o; o >>= 1) s += __shfl_xor(s, o, 32);
  float mean = s * (1.0f / 128.0f);
  float vs = 0.0f;
#pragma unroll
  for (int i = 0; i < 4; ++i) { float d0 = v[i] - mean; vs += d0 * d0; }
#pragma unroll
  for (int o = 16; o; o >>= 1) vs += __shfl_xor(vs, o, 32);
  float inv = rsqrtf(vs * (1.0f / 128.0f) + 1e-8f);
  float* q = out + (size_t)t * Ddim;
#pragma unroll
  for (int i = 0; i < 4; ++i) {
    int d0 = lane + 32 * i;
    q[d0] = (v[i] - mean) * inv * w[d0] + b[d0];
  }
}

// ---------------------------------------------------------------------------
// Shared fragment loader: A/B fragment rows from padded f16 LDS.
//   halves [0..7]  = K kb..kb+7, halves [8..15] = K kb+16..kb+23
//   (with kb = ksBase + (lane>=16 ? 8 : 0) this matches the ISA 16-bit layout)
// Row stride must be a multiple of 8 halves (16B alignment for b128 loads).
// ---------------------------------------------------------------------------
template <int ST>
__device__ __forceinline__ v16h load_frag(const _Float16* base, int row, int kb) {
  union { v16h h; float4 f[2]; } u;
  const _Float16* p = base + row * ST + kb;
  u.f[0] = *reinterpret_cast<const float4*>(p);
  u.f[1] = *reinterpret_cast<const float4*>(p + 16);
  return u.h;
}

// ---------------------------------------------------------------------------
// WMMA GEMM: Out[M,128] = epilogue(In[M,128] @ W[128,128]^T + bias)
//   epilogue: + resid_scale*resid, ReLU, * rowmask[row], * out_scale,
//             store f32 (OutF) or f16 (OutH).
// 256 threads (8 waves); 64 M-rows/block; wave w owns N-tile [16w,16w+16).
// 16 WMMAs/wave (4 K-steps x 4 M-subtiles), B-fragment reused across M.
// Weights arrive pre-converted f16 -> staging is pure b128 copy.
// ---------------------------------------------------------------------------
#define GST 136   // LDS row stride (halves): 272B, conflict-free b128 frags

__global__ void gemm128_kernel(const float* __restrict__ In,
                               const _Float16* __restrict__ W16,
                               const float* __restrict__ bias,
                               const float* __restrict__ resid,
                               float resid_scale,
                               const float* __restrict__ rowmask,
                               int relu, float out_scale,
                               float* __restrict__ OutF,
                               _Float16* __restrict__ OutH) {
  __shared__ _Float16 sW[128 * GST];   // W[n][k] as f16
  __shared__ _Float16 sIn[64 * GST];   // 64-row input tile as f16

  const int tid = threadIdx.x;
  const int m0  = blockIdx.x * 64;

  // Stage weights: straight 16B copies (pre-converted f16).
  for (int g = tid; g < 2048; g += 256) {
    int n = g >> 4, ko = (g & 15) * 8;
    *reinterpret_cast<float4*>(&sW[n * GST + ko]) =
        *reinterpret_cast<const float4*>(&W16[n * 128 + ko]);
  }
  // Stage activations: float4 loads, convert, b64 LDS stores.
  for (int g = tid; g < 2048; g += 256) {
    int r = g >> 5, ko = (g & 31) * 4;
    float4 f = *reinterpret_cast<const float4*>(
        &In[(size_t)(m0 + r) * Ddim + ko]);
    union { _Float16 h[4]; uint2 u; } cv;
    cv.h[0] = (_Float16)f.x; cv.h[1] = (_Float16)f.y;
    cv.h[2] = (_Float16)f.z; cv.h[3] = (_Float16)f.w;
    *reinterpret_cast<uint2*>(&sIn[r * GST + ko]) = cv.u;
  }
  __syncthreads();

  const int wave = tid >> 5;          // N-tile index 0..7
  const int lane = tid & 31;
  const int lrow = lane & 15;
  const int khalf = lane >> 4;

  v8f acc[4] = {};
#pragma unroll
  for (int ks = 0; ks < 4; ++ks) {
    int kb = ks * 32 + khalf * 8;
    v16h bf = load_frag<GST>(sW, wave * 16 + lrow, kb);   // B(k,n)=W[n][k]
#pragma unroll
    for (int mt = 0; mt < 4; ++mt) {
      v16h af = load_frag<GST>(sIn, mt * 16 + lrow, kb);
      acc[mt] = wmma_f16(af, bf, acc[mt]);
    }
  }

  const int n = wave * 16 + lrow;
  const int mhalf = khalf * 8;
  const float bn = bias[n];
#pragma unroll
  for (int mt = 0; mt < 4; ++mt) {
#pragma unroll
    for (int r = 0; r < 8; ++r) {
      int m = m0 + mt * 16 + mhalf + r;
      float v = acc[mt][r] + bn;
      if (resid)   v += resid_scale * resid[(size_t)m * Ddim + n];
      if (relu)    v = v > 0.0f ? v : 0.0f;
      if (rowmask) v *= rowmask[m];
      v *= out_scale;
      if (OutH) OutH[(size_t)m * Ddim + n] = (_Float16)v;
      else      OutF[(size_t)m * Ddim + n] = v;
    }
  }
}

// ---------------------------------------------------------------------------
// WMMA flash attention: one wave32 per (b, h, 16-query tile).
// Q/K/V arrive as f16 (Q pre-scaled by 1/sqrt(DH) in its GEMM epilogue), so
// staging is pure vectorized copy. Two-pass softmax (row-max, then exp/sum +
// P@V) keeps VGPR pressure low. Causal + timeline masking with -1e9.
// ---------------------------------------------------------------------------
#define QST 72    // Q/K tile stride (64 d, padded)
#define PST 40    // P tile stride (32 k, padded)
#define VST 40    // V^T tile stride (32 k, padded)

__global__ void attn_wmma_kernel(const _Float16* __restrict__ Q,
                                 const _Float16* __restrict__ K,
                                 const _Float16* __restrict__ V,
                                 const float* __restrict__ tmaskf,
                                 float* __restrict__ Out) {
  __shared__ _Float16 sQ[16 * QST];
  __shared__ _Float16 sK[16 * QST];
  __shared__ _Float16 sP[16 * PST];
  __shared__ _Float16 sVt[64 * VST];   // V transposed: [d][k_local]

  const int lane = threadIdx.x;
  int idx = blockIdx.x;
  const int qt = idx % NQT;
  const int h  = (idx / NQT) % Hdim;
  const int b  = idx / (NQT * Hdim);
  const int rowbase = b * Ldim;
  const int dbase   = h * DHdim;

  const int lrow  = lane & 15;
  const int khalf = lane >> 4;
  const int qrow0 = qt * 16 + 8 * khalf;   // + r gives this lane's C rows

  const uint4 z4 = make_uint4(0, 0, 0, 0);

  // Stage Q tile: 16 rows x 8 uint4 (16B) copies.
  for (int g = lane; g < 128; g += 32) {
    int qr = g >> 3, ko = (g & 7) * 8;
    int gq = qt * 16 + qr;
    uint4 u = (gq < Ldim)
        ? *reinterpret_cast<const uint4*>(
              &Q[(size_t)(rowbase + gq) * Ddim + dbase + ko])
        : z4;
    *reinterpret_cast<uint4*>(&sQ[qr * QST + ko]) = u;
  }
  __syncthreads();
  v16h aq0 = load_frag<QST>(sQ, lrow, khalf * 8);
  v16h aq1 = load_frag<QST>(sQ, lrow, 32 + khalf * 8);

  const int nkt = qt + 1;   // causal: key tiles 0..qt

  // ---- pass 1: masked row max ----
  float rowmax[8];
#pragma unroll
  for (int r = 0; r < 8; ++r) rowmax[r] = -3.0e38f;

  for (int kt = 0; kt < nkt; ++kt) {
    for (int g = lane; g < 128; g += 32) {
      int kr = g >> 3, ko = (g & 7) * 8;
      int kk = kt * 16 + kr;
      uint4 u = (kk < Ldim)
          ? *reinterpret_cast<const uint4*>(
                &K[(size_t)(rowbase + kk) * Ddim + dbase + ko])
          : z4;
      *reinterpret_cast<uint4*>(&sK[kr * QST + ko]) = u;
    }
    __syncthreads();
    v16h bk0 = load_frag<QST>(sK, lrow, khalf * 8);
    v16h bk1 = load_frag<QST>(sK, lrow, 32 + khalf * 8);
    v8f c = {};
    c = wmma_f16(aq0, bk0, c);
    c = wmma_f16(aq1, bk1, c);
    int kkg = kt * 16 + lrow;
    float tmv = (kkg < Ldim) ? tmaskf[rowbase + kkg] : 0.0f;
#pragma unroll
    for (int r = 0; r < 8; ++r) {
      bool valid = (kkg <= qrow0 + r) && (tmv != 0.0f);
      float s = valid ? c[r] : -1.0e9f;
      rowmax[r] = fmaxf(rowmax[r], s);
    }
    __syncthreads();
  }
#pragma unroll
  for (int r = 0; r < 8; ++r)
#pragma unroll
    for (int o = 8; o; o >>= 1)
      rowmax[r] = fmaxf(rowmax[r], __shfl_xor(rowmax[r], o, 32));

  // ---- pass 2: exp / sum + P@V ----
  float rowsum[8];
#pragma unroll
  for (int r = 0; r < 8; ++r) rowsum[r] = 0.0f;
  v8f acc[4] = {};

  const int nchunk = (nkt + 1) >> 1;   // 32-key chunks
  for (int ch = 0; ch < nchunk; ++ch) {
    for (int hf = 0; hf < 2; ++hf) {
      int kt = 2 * ch + hf;
      int colbase = hf * 16;
      if (kt < nkt) {
        for (int g = lane; g < 128; g += 32) {
          int kr = g >> 3, ko = (g & 7) * 8;
          int kk = kt * 16 + kr;
          uint4 u = (kk < Ldim)
              ? *reinterpret_cast<const uint4*>(
                    &K[(size_t)(rowbase + kk) * Ddim + dbase + ko])
              : z4;
          *reinterpret_cast<uint4*>(&sK[kr * QST + ko]) = u;
        }
        __syncthreads();
        v16h bk0 = load_frag<QST>(sK, lrow, khalf * 8);
        v16h bk1 = load_frag<QST>(sK, lrow, 32 + khalf * 8);
        v8f c = {};
        c = wmma_f16(aq0, bk0, c);
        c = wmma_f16(aq1, bk1, c);
        int kkg = kt * 16 + lrow;
        float tmv = (kkg < Ldim) ? tmaskf[rowbase + kkg] : 0.0f;
#pragma unroll
        for (int r = 0; r < 8; ++r) {
          bool valid = (kkg <= qrow0 + r) && (tmv != 0.0f);
          float s = valid ? c[r] : -1.0e9f;
          float p = __expf(s - rowmax[r]);
          rowsum[r] += p;
          sP[(r + 8 * khalf) * PST + colbase + lrow] = (_Float16)p;
        }
      } else {
#pragma unroll
        for (int r = 0; r < 8; ++r)
          sP[(r + 8 * khalf) * PST + colbase + lrow] = (_Float16)0.0f;
      }
      __syncthreads();
    }
    // Stage V transposed for this 32-key chunk: sVt[d][k_local].
    for (int i = lane; i < 64 * 32; i += 32) {
      int d = i >> 5, kl = i & 31;
      int kk = ch * 32 + kl;
      _Float16 v = (kk < Ldim)
          ? V[(size_t)(rowbase + kk) * Ddim + dbase + d]
          : (_Float16)0.0f;
      sVt[d * VST + kl] = v;
    }
    __syncthreads();
    v16h ap = load_frag<PST>(sP, lrow, khalf * 8);
#pragma unroll
    for (int j = 0; j < 4; ++j) {
      v16h bv = load_frag<VST>(sVt, j * 16 + lrow, khalf * 8);
      acc[j] = wmma_f16(ap, bv, acc[j]);
    }
    __syncthreads();
  }

#pragma unroll
  for (int r = 0; r < 8; ++r)
#pragma unroll
    for (int o = 8; o; o >>= 1)
      rowsum[r] += __shfl_xor(rowsum[r], o, 32);

#pragma unroll
  for (int j = 0; j < 4; ++j)
#pragma unroll
    for (int r = 0; r < 8; ++r) {
      int q = qrow0 + r;
      if (q < Ldim)
        Out[(size_t)(rowbase + q) * Ddim + dbase + j * 16 + lrow] =
            acc[j][r] / rowsum[r];
    }
}

// ---------------------------------------------------------------------------
// Host orchestration.
// ---------------------------------------------------------------------------
namespace {

struct EncParams {
  // per block, sorted-key order:
  // 0:W1 1:W2 2:Wk 3:Wo 4:Wq 5:Wv 6:b1 7:b2 8:bk 9:bo 10:bq 11:bv
  // 12:ln1_b 13:ln1_w 14:ln2_b 15:ln2_w
  const float* blk[2][16];
  const float *ln_b, *ln_w, *pos_emb;
};

EncParams parse_enc(void* const* d_in, int base) {
  EncParams E;
  int i = base;
  for (int bl = 0; bl < 2; ++bl)
    for (int j = 0; j < 16; ++j) E.blk[bl][j] = (const float*)d_in[i++];
  E.ln_b    = (const float*)d_in[i++];
  E.ln_w    = (const float*)d_in[i++];
  E.pos_emb = (const float*)d_in[i++];
  return E;
}

void run_encoder(const int* ids, const int* pos, const float* emb,
                 const EncParams& E, float* ws, float* out,
                 hipStream_t stream) {
  // Workspace layout (ws is float*):
  float*     xb = ws;              // running activation x      (f32, NELT)
  float*     qn = ws + 1 * NELT;   // LN output (Qn / y)        (f32, NELT)
  float*     ab = ws + 2 * NELT;   // attn out / FFN hidden     (f32, NELT)
  _Float16*  qh = (_Float16*)(ws + 3 * NELT);   // Q proj (f16, NELT)
  _Float16*  kh = qh + NELT;                    // K proj (f16, NELT)
  _Float16*  vh = kh + NELT;                    // V proj (f16, NELT)
  float*     tm = (float*)(vh + NELT);          // timeline mask [NTOK]
  _Float16*  w16 = (_Float16*)(tm + NTOK);      // 12 x 16384 f16 weights

  dim3 lnBlk(32, 8);
  int  lnGrid   = NTOK / 8;
  int  gemmGrid = NTOK / 64;

  // Pre-convert this encoder's 12 weight matrices to f16 (once per call).
  for (int bl = 0; bl < 2; ++bl)
    for (int j = 0; j < 6; ++j)
      wcvt_kernel<<<WMAT / 1024, 256, 0, stream>>>(
          E.blk[bl][j], w16 + (size_t)(bl * 6 + j) * WMAT);

  embed_kernel<<<NTOK, Ddim, 0, stream>>>(ids, pos, emb, E.pos_emb, xb, tm);

  for (int bl = 0; bl < 2; ++bl) {
    const float* const* P = E.blk[bl];
    const _Float16* w = w16 + (size_t)bl * 6 * WMAT;
    const _Float16 *W1 = w, *W2 = w + WMAT, *Wk = w + 2 * WMAT,
                   *Wo = w + 3 * WMAT, *Wq = w + 4 * WMAT, *Wv = w + 5 * WMAT;
    // Qn = LN1(x)
    ln_kernel<<<lnGrid, lnBlk, 0, stream>>>(xb, P[13], P[12], qn, NTOK);
    // Q (pre-scaled by 1/8, f16 out); K,V from pre-LN x (f16 out)
    gemm128_kernel<<<gemmGrid, 256, 0, stream>>>(qn, Wq, P[10], nullptr, 0.f, nullptr, 0, 0.125f, nullptr, qh);
    gemm128_kernel<<<gemmGrid, 256, 0, stream>>>(xb, Wk, P[8],  nullptr, 0.f, nullptr, 0, 1.0f,   nullptr, kh);
    gemm128_kernel<<<gemmGrid, 256, 0, stream>>>(xb, Wv, P[11], nullptr, 0.f, nullptr, 0, 1.0f,   nullptr, vh);
    // causal masked softmax attention (WMMA flash)
    attn_wmma_kernel<<<Bdim * Hdim * NQT, 32, 0, stream>>>(qh, kh, vh, tm, ab);
    // x = Qn + attn@Wo^T + bo
    gemm128_kernel<<<gemmGrid, 256, 0, stream>>>(ab, Wo, P[9], qn, 1.0f, nullptr, 0, 1.0f, xb, nullptr);
    // y = LN2(x)
    ln_kernel<<<lnGrid, lnBlk, 0, stream>>>(xb, P[15], P[14], qn, NTOK);
    // h = relu(y@W1^T+b1)  -> ab (attn buffer is dead now)
    gemm128_kernel<<<gemmGrid, 256, 0, stream>>>(qn, W1, P[6], nullptr, 0.f, nullptr, 1, 1.0f, ab, nullptr);
    // x = (h@W2^T + b2 + 2*y) * tmask   (double residual, faithful)
    gemm128_kernel<<<gemmGrid, 256, 0, stream>>>(ab, W2, P[7], qn, 2.0f, tm, 0, 1.0f, xb, nullptr);
  }
  // final LN straight into the output slice
  ln_kernel<<<lnGrid, lnBlk, 0, stream>>>(xb, E.ln_w, E.ln_b, out, NTOK);
}

} // namespace

extern "C" void kernel_launch(void* const* d_in, const int* in_sizes, int n_in,
                              void* d_out, int out_size, void* d_ws, size_t ws_size,
                              hipStream_t stream) {
  (void)in_sizes; (void)n_in; (void)out_size; (void)ws_size;
  // d_in layout: 0-2 ids (o,x,y), 3-5 positions, 6-8 emb tables,
  // then enc/enc_X/enc_Y as 35 leaves each (sorted-key pytree order).
  float* out = (float*)d_out;
  float* ws  = (float*)d_ws;
  for (int e = 0; e < 3; ++e) {
    EncParams E = parse_enc(d_in, 9 + 35 * e);
    run_encoder((const int*)d_in[e], (const int*)d_in[3 + e],
                (const float*)d_in[6 + e], E, ws, out + (size_t)e * NELT,
                stream);
  }
}
